// CrossAttentionModule_1013612281948
// MI455X (gfx1250) — matrix operand order repaired and checked by reading, not compiled, not run on hardware
//
#include <hip/hip_runtime.h>
#include <hip/hip_bf16.h>

typedef __attribute__((ext_vector_type(16))) _Float16 v16h;
typedef __attribute__((ext_vector_type(8)))  _Float16 v8h;
typedef __attribute__((ext_vector_type(8)))  float    v8f;

static constexpr int kB = 8, kC = 256, kHW = 4096, kCQK = 32;

__device__ __forceinline__ v8f wmma_f16(const v16h& a, const v16h& b,
                                        const v8f& c) {
  return __builtin_amdgcn_wmma_f32_16x16x32_f16(false, a, false, b,
                                                (short)0, c, false, false);
}

// ---------------------------------------------------------------------------
// Kernel 1: 1x1-conv projections, fully on the WMMA pipe (memory-bound).
// Computes D[o][n] = W_tile(16o x 32c) x X_tile(32c x 16n) per k-step.
//   A = weight rows  (contiguous f32 -> f16, b128 loads)
//   B = feature cols (dword gathers, coalesced across the 16 pixel lanes)
// Block = 128 threads (4 waves) per 16-pixel tile. Every wave runs the SAME
// instruction stream (EXEC stays all-1s around WMMA): 4 V-tiles (64 ch) plus
// one Q-or-K tile chosen by wave-uniform pointer selection.
// Outputs: Qh/Kh [B][HW][32] row-major f16, Vh [B][C][HW] channel-major f16.
// ---------------------------------------------------------------------------
__global__ __launch_bounds__(128) void proj_qkv_wmma(
    const float* __restrict__ f1, const float* __restrict__ f2,
    const float* __restrict__ f3,
    const float* __restrict__ wq, const float* __restrict__ bq,
    const float* __restrict__ wk, const float* __restrict__ bk,
    const float* __restrict__ wv, const float* __restrict__ bv,
    _Float16* __restrict__ Qh, _Float16* __restrict__ Kh,
    _Float16* __restrict__ Vh)
{
  const int lane  = threadIdx.x & 31;
  const int wave  = threadIdx.x >> 5;
  const int b     = blockIdx.x >> 8;           // / 256 pixel tiles per batch
  const int n0    = (blockIdx.x & 255) * 16;   // first pixel of tile
  const int half  = lane >> 4;
  const int l15   = lane & 15;
  const int khalf = half * 8;

  const float* F3  = f3 + (size_t)b * kC * kHW;
  const float* Fqk = ((wave < 2) ? f1 : f2) + (size_t)b * kC * kHW;
  const float* Wqk = (wave < 2) ? wq : wk;
  const float* Bqk = (wave < 2) ? bq : bk;
  const int    qko   = (wave & 1) * 16;        // Q/K o-tile base (0 or 16)
  const int    vbase = wave * 64;              // V channel slice base

  v8f vacc[4];
  v8f qkacc;
#pragma unroll
  for (int t = 0; t < 4; ++t)
#pragma unroll
    for (int i = 0; i < 8; ++i) vacc[t][i] = 0.0f;
#pragma unroll
  for (int i = 0; i < 8; ++i) qkacc[i] = 0.0f;

  for (int ks = 0; ks < kC / 32; ++ks) {
    const int c0 = ks * 32;

    // --- B tiles: feature columns (K = channels, strided, lane-coalesced) ---
    v16h b3, bqk_t;
    {
      const float* X3 = F3  + (size_t)(c0 + half * 16) * kHW + n0 + l15;
      const float* Xq = Fqk + (size_t)(c0 + half * 16) * kHW + n0 + l15;
#pragma unroll
      for (int i = 0; i < 16; ++i) {
        b3[i]    = (_Float16)X3[(size_t)i * kHW];
        bqk_t[i] = (_Float16)Xq[(size_t)i * kHW];
      }
    }

    // --- A tiles: weight rows (contiguous f32 -> f16) ---
    auto loadA = [&](const float* W, int obase) -> v16h {
      const float* Wp = W + (size_t)(obase + l15) * kC + c0;
      v16h a;
#pragma unroll
      for (int i = 0; i < 8; ++i) {
        a[i]     = (_Float16)Wp[half * 16 + i];
        a[i + 8] = (_Float16)Wp[half * 16 + 8 + i];
      }
      return a;
    };

#pragma unroll
    for (int t = 0; t < 4; ++t)
      vacc[t] = wmma_f16(loadA(wv, vbase + t * 16), b3, vacc[t]);
    qkacc = wmma_f16(loadA(Wqk, qko), bqk_t, qkacc);
  }

  // --- Q/K store: row-major [n][32], one packed 16B store per lane ---
  {
    _Float16* Dst = ((wave < 2) ? Qh : Kh)
                    + ((size_t)b * kHW + n0 + l15) * kCQK + qko + half * 8;
    v8h pack;
#pragma unroll
    for (int i = 0; i < 8; ++i)
      pack[i] = (_Float16)(qkacc[i] + Bqk[qko + half * 8 + i]);
    *(v8h*)Dst = pack;
  }

  // --- V store: channel-major [c][n] (matches attention GEMM2 A layout) ---
  _Float16* Vo = Vh + (size_t)b * kC * kHW + n0 + l15;
#pragma unroll
  for (int t = 0; t < 4; ++t) {
    const int ob = vbase + t * 16 + half * 8;
#pragma unroll
    for (int i = 0; i < 8; ++i)
      Vo[(size_t)(ob + i) * kHW] = (_Float16)(vacc[t][i] + bv[ob + i]);
  }
}

// ---------------------------------------------------------------------------
// Kernel 2: flash-attention + fused output, all matrix math in
// v_wmma_f32_16x16x32_f16. Block = 128 threads (4 waves), one 16-query tile
// per block; each wave owns a 64-channel output slice (4 accumulator tiles).
// Loop over keys in blocks of 32:
//   GEMM1: St(16key x 16q) = K_tile(16x32) x Q^T(32x16)   [x2 key sub-tiles]
//   online softmax per query column (in-lane + one shfl_xor(16))
//   GEMM2: Out^T(16c x 16q) += V^T_tile(16x32) x P(32x16) [x4 channel tiles]
// P chains from GEMM1's D layout into GEMM2's B layout with 16 shfl_xor ops.
// ---------------------------------------------------------------------------
__global__ __launch_bounds__(128) void flash_attn(
    const _Float16* __restrict__ Qh, const _Float16* __restrict__ Kh,
    const _Float16* __restrict__ Vh, float* __restrict__ out)
{
  const int lane  = threadIdx.x & 31;
  const int wave  = threadIdx.x >> 5;
  const int b     = blockIdx.x >> 8;           // / 256 query tiles per batch
  const int n0    = (blockIdx.x & 255) * 16;   // first query of tile
  const int half  = lane >> 4;                 // 0 | 1
  const int l15   = lane & 15;
  const int khalf = half * 8;

  // B operand of GEMM1: Q columns. lane<16 holds ck 0..15 of query n0+lane,
  // lane>=16 holds ck 16..31 of query n0+lane-16. Loop-invariant.
  const v16h qb = *(const v16h*)(Qh + ((size_t)b * kHW + (n0 + l15)) * kCQK
                                 + half * 16);

  v8f acc[4];
#pragma unroll
  for (int tI = 0; tI < 4; ++tI)
#pragma unroll
    for (int i = 0; i < 8; ++i) acc[tI][i] = 0.0f;

  float mrun = -__builtin_inff();
  float lrun = 0.0f;
  const int cb = wave * 64;
  const _Float16* Kbase = Kh + (size_t)b * kHW * kCQK;
  const _Float16* Vbase = Vh + (size_t)b * kC * kHW;

  for (int m0 = 0; m0 < kHW; m0 += 32) {
    // --- GEMM1 A operands: two 16-key tiles (rows = keys, contiguous 16B) ---
    const _Float16* Kp0 = Kbase + (size_t)(m0 + l15) * kCQK;
    const _Float16* Kp1 = Kbase + (size_t)(m0 + 16 + l15) * kCQK;
    // prefetch next key block (global_prefetch_b8)
    __builtin_prefetch((const void*)(Kp0 + 32 * kCQK), 0, 1);
    v16h ka0, ka1;
    {
      v8h a = *(const v8h*)(Kp0 + half * 16);
      v8h c = *(const v8h*)(Kp0 + half * 16 + 8);
      v8h d = *(const v8h*)(Kp1 + half * 16);
      v8h e = *(const v8h*)(Kp1 + half * 16 + 8);
#pragma unroll
      for (int i = 0; i < 8; ++i) {
        ka0[i] = a[i]; ka0[i + 8] = c[i];
        ka1[i] = d[i]; ka1[i + 8] = e[i];
      }
    }
    const v8f z = {};
    v8f st0 = wmma_f16(ka0, qb, z);
    v8f st1 = wmma_f16(ka1, qb, z);

    // --- online softmax per query column ---
    float bm = st0[0];
#pragma unroll
    for (int i = 0; i < 8; ++i) {
      bm = fmaxf(bm, st0[i]);
      bm = fmaxf(bm, st1[i]);
    }
    bm = fmaxf(bm, __shfl_xor(bm, 16));
    const float mnew = fmaxf(mrun, bm);
    const float corr = __expf(mrun - mnew);

    float p0[8], p1[8], ls = 0.0f;
#pragma unroll
    for (int i = 0; i < 8; ++i) {
      p0[i] = __expf(st0[i] - mnew);
      p1[i] = __expf(st1[i] - mnew);
      ls += p0[i] + p1[i];
    }
    ls += __shfl_xor(ls, 16);
    lrun = lrun * corr + ls;
    mrun = mnew;
#pragma unroll
    for (int tI = 0; tI < 4; ++tI)
#pragma unroll
      for (int i = 0; i < 8; ++i) acc[tI][i] *= corr;

    // --- chain P into GEMM2 B operand (columns = queries, K = 32 keys) ---
    v16h pb;
#pragma unroll
    for (int i = 0; i < 8; ++i) {
      const float o0 = __shfl_xor(p0[i], 16);   // keys 8..15 (from pair lane)
      const float o1 = __shfl_xor(p1[i], 16);   // keys 16..23 (from pair lane)
      if (half == 0) { pb[i] = (_Float16)p0[i]; pb[i + 8] = (_Float16)o0; }
      else           { pb[i] = (_Float16)o1;   pb[i + 8] = (_Float16)p1[i]; }
    }

    // --- GEMM2: 4 channel tiles, A rows = channels of V^T (contiguous) ---
#pragma unroll
    for (int tI = 0; tI < 4; ++tI) {
      const _Float16* Vp =
          Vbase + (size_t)(cb + tI * 16 + l15) * kHW + m0 + half * 16;
      __builtin_prefetch((const void*)(Vp + 32), 0, 1);  // next key block
      v8h vlo = *(const v8h*)Vp;
      v8h vhi = *(const v8h*)(Vp + 8);
      v16h va;
#pragma unroll
      for (int i = 0; i < 8; ++i) { va[i] = vlo[i]; va[i + 8] = vhi[i]; }
      acc[tI] = wmma_f16(va, pb, acc[tI]);
    }
  }

  // --- normalize and store: out[b][c][n], c from VGPR index, n = l15 ---
  const float rinv = 1.0f / lrun;
  float* Op = out + (size_t)b * kC * kHW + (n0 + l15);
#pragma unroll
  for (int tI = 0; tI < 4; ++tI) {
    const int rb = cb + tI * 16 + half * 8;
#pragma unroll
    for (int i = 0; i < 8; ++i)
      Op[(size_t)(rb + i) * kHW] = acc[tI][i] * rinv;
  }
}

// ---------------------------------------------------------------------------
extern "C" void kernel_launch(void* const* d_in, const int* in_sizes, int n_in,
                              void* d_out, int out_size, void* d_ws,
                              size_t ws_size, hipStream_t stream)
{
  (void)in_sizes; (void)n_in; (void)out_size; (void)ws_size;

  const float* f1 = (const float*)d_in[0];
  const float* f2 = (const float*)d_in[1];
  const float* f3 = (const float*)d_in[2];
  const float* wq = (const float*)d_in[3];
  const float* bq = (const float*)d_in[4];
  const float* wk = (const float*)d_in[5];
  const float* bk = (const float*)d_in[6];
  const float* wv = (const float*)d_in[7];
  const float* bv = (const float*)d_in[8];

  _Float16* Qh = (_Float16*)d_ws;                       // 2 MB
  _Float16* Kh = Qh + (size_t)kB * kHW * kCQK;          // 2 MB
  _Float16* Vh = Kh + (size_t)kB * kHW * kCQK;          // 16 MB

  proj_qkv_wmma<<<kB * (kHW / 16), 128, 0, stream>>>(
      f1, f2, f3, wq, bq, wk, bk, wv, bv, Qh, Kh, Vh);

  flash_attn<<<kB * (kHW / 16), 128, 0, stream>>>(
      Qh, Kh, Vh, (float*)d_out);
}
